// PointNet2_24120536335105
// MI455X (gfx1250) — matrix operand order repaired and checked by reading, not compiled, not run on hardware
//
#include <hip/hip_runtime.h>
#include <hip/hip_bf16.h>

typedef __attribute__((ext_vector_type(16))) _Float16 v16h;
typedef __attribute__((ext_vector_type(8)))  float    v8f;

#define BATCH 8
#define NPTS  16384
#define NCENT 512

struct LayerParams { const float *W, *b, *g, *be; };
struct MLPParams   { LayerParams L[3]; };

// ---------------------------------------------------------------------------
// FPS: one block per batch. dist[] lives in workspace (stays hot in L2).
// Sequential 512-step argmax with first-index tie-break (matches jnp.argmax).
// ---------------------------------------------------------------------------
__global__ __launch_bounds__(256) void fps_kernel(const float* __restrict__ xyz,
                                                  int* __restrict__ cent,
                                                  float* __restrict__ distws) {
  const int b = blockIdx.x, tid = threadIdx.x;
  const float* px = xyz + (size_t)b * 3 * NPTS;
  float* dist = distws + (size_t)b * NPTS;
  for (int i = tid; i < NPTS; i += 256) dist[i] = 1e10f;
  __shared__ float rmax[256];
  __shared__ int   ridx[256];
  __shared__ int   scur;
  if (tid == 0) scur = 0;
  __syncthreads();
  for (int s = 0; s < NCENT; ++s) {
    const int cur = scur;
    if (tid == 0) cent[b * NCENT + s] = cur;
    const float cx = px[cur], cy = px[NPTS + cur], cz = px[2 * NPTS + cur];
    float best = -1.0f; int bidx = 0;
    for (int i = tid; i < NPTS; i += 256) {
      float dx = px[i] - cx, dy = px[NPTS + i] - cy, dz = px[2 * NPTS + i] - cz;
      float d = dx * dx + dy * dy + dz * dz;
      d = fminf(dist[i], d);
      dist[i] = d;
      if (d > best) { best = d; bidx = i; }
    }
    rmax[tid] = best; ridx[tid] = bidx;
    __syncthreads();
    for (int off = 128; off > 0; off >>= 1) {
      if (tid < off) {
        float o = rmax[tid + off]; int oi = ridx[tid + off];
        if (o > rmax[tid] || (o == rmax[tid] && oi < ridx[tid])) {
          rmax[tid] = o; ridx[tid] = oi;
        }
      }
      __syncthreads();
    }
    if (tid == 0) scur = ridx[0];
    __syncthreads();
  }
}

// ---------------------------------------------------------------------------
// Gather centroid coordinates: writes new_xyz output (B,3,S) and a (B*S,3)
// workspace copy for the ball-query / grouping kernels.
// ---------------------------------------------------------------------------
__global__ __launch_bounds__(256) void gather_kernel(const float* __restrict__ xyz,
                                                     const int* __restrict__ cent,
                                                     float* __restrict__ out_nx,
                                                     float* __restrict__ nxyz_ws) {
  int t = blockIdx.x * 256 + threadIdx.x;
  if (t >= BATCH * NCENT) return;
  int b = t / NCENT, s = t % NCENT;
  int idx = cent[t];
  for (int c = 0; c < 3; ++c) {
    float v = xyz[((size_t)b * 3 + c) * NPTS + idx];
    out_nx[((size_t)b * 3 + c) * NCENT + s] = v;
    nxyz_ws[(size_t)t * 3 + c] = v;
  }
}

// ---------------------------------------------------------------------------
// Ball query: one wave32 per centroid. Ballot-compact the first K in-range
// indices in ascending index order (== reference top_k on indices), then
// fill the tail with the first hit.
// ---------------------------------------------------------------------------
template <int K>
__global__ __launch_bounds__(256) void ballquery_kernel(const float* __restrict__ xyz,
                                                        const float* __restrict__ nxyz,
                                                        int* __restrict__ gi,
                                                        float r2) {
  const int wave = threadIdx.x >> 5, lane = threadIdx.x & 31;
  const int cid = blockIdx.x * 8 + wave;
  const int b = cid / NCENT;
  const float cx = nxyz[cid * 3 + 0], cy = nxyz[cid * 3 + 1], cz = nxyz[cid * 3 + 2];
  const float* px = xyz + (size_t)b * 3 * NPTS;
  int count = 0;
  int first = -1;
  for (int base = 0; base < NPTS && count < K; base += 32) {
    const int i = base + lane;
    float dx = px[i] - cx, dy = px[NPTS + i] - cy, dz = px[2 * NPTS + i] - cz;
    bool in = (dx * dx + dy * dy + dz * dz) <= r2;
    unsigned m = __builtin_amdgcn_ballot_w32(in);
    if (first < 0 && m) first = base + __ffs((int)m) - 1;
    int pos = __popc(m & ((1u << lane) - 1u));
    if (in && (count + pos) < K) gi[(size_t)cid * K + count + pos] = i;
    count += __popc(m);
  }
  // count >= 1 always (centroid is inside its own ball). Fill remaining slots.
  for (int j = count + lane; j < K; j += 32) gi[(size_t)cid * K + j] = first;
}

// ---------------------------------------------------------------------------
// Grouped MLP + LayerNorm + ReLU + max-pool using v_wmma_f32_16x16x32_f16.
// Block = 256 threads (8 waves) handles 64 rows = G groups * K samples.
// Wave w: row-tile mt = w>>1 (16 rows), column half nh = w&1.
// ---------------------------------------------------------------------------
template <int KS, int G, int CBASE, int C0, int C1, int C2>
__global__ __launch_bounds__(256) void mlp_kernel(const float* __restrict__ xyz,
                                                  const float* __restrict__ points,
                                                  const float* __restrict__ nxyz,
                                                  const int* __restrict__ gi,
                                                  MLPParams mp,
                                                  float* __restrict__ out_np) {
  constexpr int ROWS = G * KS;           // always 64
  __shared__ _Float16 Xlds[64 * 128];    // activations (f16), stride = padded cin
  __shared__ _Float16 Wlds[32 * 128];    // current 32-row k-chunk of W (f16)
  __shared__ float    Ylds[64 * 128];    // layer output (f32)

  const int tid = threadIdx.x;
  const int b = blockIdx.x / (NCENT / G);
  const int s0 = (blockIdx.x % (NCENT / G)) * G;

  // ---- gather input features: [points(3) | xyz - new_xyz(3)] padded to 32 ch
  for (int e = tid; e < ROWS * 32; e += 256) {
    int row = e >> 5, c = e & 31;
    int g = row / KS, k = row % KS;
    int s = s0 + g;
    float v = 0.0f;
    if (c < 6) {
      int idx = gi[((size_t)(b * NCENT + s)) * KS + k];
      if (c < 3) v = points[((size_t)b * 3 + c) * NPTS + idx];
      else       v = xyz[((size_t)b * 3 + (c - 3)) * NPTS + idx]
                   - nxyz[(size_t)(b * NCENT + s) * 3 + (c - 3)];
    }
    Xlds[row * 32 + c] = (_Float16)v;
  }
  __syncthreads();

  const int cins[3]  = {32, C0, C1};   // padded input channels (multiples of 32)
  const int rcins[3] = {6,  C0, C1};   // real input channels
  const int couts[3] = {C0, C1, C2};

  const int wave = tid >> 5, lane = tid & 31;
  const int mt = wave >> 1, nh = wave & 1;
  const int koffs[8] = {0, 2, 4, 6, 16, 18, 20, 22};

  for (int layer = 0; layer < 3; ++layer) {
    const int cin = cins[layer], rcin = rcins[layer], cout = couts[layer];
    const LayerParams P = mp.L[layer];
    const int ntilesHalf = cout >> 5;    // 16-col tiles per wave (1..4)

    v8f acc[4] = {};
    for (int kk = 0; kk < cin; kk += 32) {
      __syncthreads();
      // stage W rows [kk, kk+32) as f16, zero-padding rows beyond rcin
      for (int e = tid; e < 32 * cout; e += 256) {
        int kr = e / cout, co = e % cout;
        int ci = kk + kr;
        float w = (ci < rcin) ? P.W[(size_t)ci * cout + co] : 0.0f;
        Wlds[kr * cout + co] = (_Float16)w;
      }
      __syncthreads();

      // A fragment: 16x32 f16 per ISA lane map
      v16h a;
      const int arow = mt * 16 + (lane & 15);
      const int kb = (lane >> 4) << 3;   // 0 or 8
#pragma unroll
      for (int j = 0; j < 16; ++j) {
        int v = j >> 1;
        a[j] = Xlds[arow * cin + kk + kb + koffs[v] + (j & 1)];
      }

      for (int i = 0; i < ntilesHalf; ++i) {
        const int nt = nh * ntilesHalf + i;
        v16h bf;
#pragma unroll
        for (int j = 0; j < 16; ++j)
          bf[j] = Wlds[(kb + koffs[j >> 1] + (j & 1)) * cout + nt * 16 + (lane & 15)];   // B: k = table order (same as A), n = lane & 15
        acc[i] = __builtin_amdgcn_wmma_f32_16x16x32_f16(
            false, a, false, bf, (short)0, acc[i], false, false);
      }
    }
    __syncthreads();

    // scatter D tiles to Ylds per ISA C/D lane map
    for (int i = 0; i < ntilesHalf; ++i) {
      const int nt = nh * ntilesHalf + i;
      const int col = nt * 16 + (lane & 15);
      const int rbase = mt * 16 + ((lane >> 4) << 3);
#pragma unroll
      for (int r = 0; r < 8; ++r)
        Ylds[(rbase + r) * cout + col] = acc[i][r];
    }
    __syncthreads();

    // bias + LayerNorm over channels + ReLU, one row per thread
    const bool last = (layer == 2);
    const int cinN = last ? 0 : cins[layer + 1];
    for (int row = tid; row < ROWS; row += 256) {
      float mu = 0.0f;
      for (int c = 0; c < cout; ++c) mu += Ylds[row * cout + c] + P.b[c];
      mu /= (float)cout;
      float var = 0.0f;
      for (int c = 0; c < cout; ++c) {
        float d = Ylds[row * cout + c] + P.b[c] - mu;
        var += d * d;
      }
      var /= (float)cout;
      const float rs = rsqrtf(var + 1e-5f);
      for (int c = 0; c < cout; ++c) {
        float y = (Ylds[row * cout + c] + P.b[c] - mu) * rs * P.g[c] + P.be[c];
        y = fmaxf(y, 0.0f);
        if (last) Ylds[row * cout + c] = y;
        else      Xlds[row * cinN + c] = (_Float16)y;   // next cin == this cout
      }
    }
    __syncthreads();
  }

  // max-pool over K samples per group, write new_points (B, 320, S)
  for (int e = tid; e < G * C2; e += 256) {
    int g = e / C2, c = e % C2;
    float m = 0.0f;  // post-ReLU values are >= 0
    for (int k = 0; k < KS; ++k) m = fmaxf(m, Ylds[(g * KS + k) * C2 + c]);
    int s = s0 + g;
    out_np[((size_t)b * 320 + CBASE + c) * NCENT + s] = m;
  }
}

// ---------------------------------------------------------------------------
extern "C" void kernel_launch(void* const* d_in, const int* in_sizes, int n_in,
                              void* d_out, int out_size, void* d_ws, size_t ws_size,
                              hipStream_t stream) {
  (void)in_sizes; (void)n_in; (void)out_size; (void)ws_size;
  const float* xyz    = (const float*)d_in[0];
  const float* points = (const float*)d_in[1];

  MLPParams mp[3];
  for (int sc = 0; sc < 3; ++sc)
    for (int l = 0; l < 3; ++l) {
      int base = 2 + 12 * sc + 4 * l;
      mp[sc].L[l].W  = (const float*)d_in[base + 0];
      mp[sc].L[l].b  = (const float*)d_in[base + 1];
      mp[sc].L[l].g  = (const float*)d_in[base + 2];
      mp[sc].L[l].be = (const float*)d_in[base + 3];
    }

  char* ws = (char*)d_ws;
  int*   cent   = (int*)ws;            ws += (size_t)BATCH * NCENT * sizeof(int);
  float* nxyz   = (float*)ws;          ws += (size_t)BATCH * NCENT * 3 * sizeof(float);
  float* distws = (float*)ws;          ws += (size_t)BATCH * NPTS * sizeof(float);
  int*   gi0    = (int*)ws;            ws += (size_t)BATCH * NCENT * 16 * sizeof(int);
  int*   gi1    = (int*)ws;            ws += (size_t)BATCH * NCENT * 32 * sizeof(int);
  int*   gi2    = (int*)ws;            ws += (size_t)BATCH * NCENT * 64 * sizeof(int);

  float* out_nx = (float*)d_out;                      // (B, 3, S)
  float* out_np = out_nx + (size_t)BATCH * 3 * NCENT; // (B, 320, S)

  fps_kernel<<<BATCH, 256, 0, stream>>>(xyz, cent, distws);
  gather_kernel<<<(BATCH * NCENT + 255) / 256, 256, 0, stream>>>(xyz, cent, out_nx, nxyz);

  ballquery_kernel<16><<<BATCH * NCENT / 8, 256, 0, stream>>>(xyz, nxyz, gi0, 0.01f);
  ballquery_kernel<32><<<BATCH * NCENT / 8, 256, 0, stream>>>(xyz, nxyz, gi1, 0.04f);
  ballquery_kernel<64><<<BATCH * NCENT / 8, 256, 0, stream>>>(xyz, nxyz, gi2, 0.16f);

  mlp_kernel<16, 4,   0, 32, 32,  64><<<BATCH * NCENT / 4, 256, 0, stream>>>(xyz, points, nxyz, gi0, mp[0], out_np);
  mlp_kernel<32, 2,  64, 64, 64, 128><<<BATCH * NCENT / 2, 256, 0, stream>>>(xyz, points, nxyz, gi1, mp[1], out_np);
  mlp_kernel<64, 1, 192, 64, 96, 128><<<BATCH * NCENT / 1, 256, 0, stream>>>(xyz, points, nxyz, gi2, mp[2], out_np);
}
